// MultiHeadAttention1_70385924047428
// MI455X (gfx1250) — matrix operand order repaired and checked by reading, not compiled, not run on hardware
//
#include <hip/hip_runtime.h>
#include <hip/hip_bf16.h>

// ---------------------------------------------------------------------------
// MI455X (gfx1250) pre-LN multi-head attention, f16 WMMA everywhere.
// B=2, L=2048, D=1024, H=16, dh=64. ~69 GFLOP over ~50MB -> compute bound;
// all four matmuls go through v_wmma_f32_16x16x32_f16 (f32 accumulate).
// GEMMs: block = 8 waves sharing one 16x32 A tile (staged to LDS with
// global_load_async_to_lds_b128 / s_wait_asynccnt), each wave owns a 16x64
// output strip (4 accumulators -> 4 WMMAs per A tile).
// ---------------------------------------------------------------------------

typedef _Float16 h16;
typedef __attribute__((ext_vector_type(16))) _Float16 v16h;
typedef __attribute__((ext_vector_type(8)))  _Float16 v8h;
typedef __attribute__((ext_vector_type(8)))  float    v8f;

#define B_   2
#define L_   2048
#define D_   1024
#define H_   16
#define DH_  64
#define D3_  3072
#define BL_  (B_ * L_)      // 4096

// ----- WMMA helpers --------------------------------------------------------

__device__ __forceinline__ v8f wmma32(v16h a, v16h b, v8f c) {
  // 8 args: (neg_a, A, neg_b, B, c_mod, C, reuse_a, reuse_b)
  return __builtin_amdgcn_wmma_f32_16x16x32_f16(false, a, false, b,
                                                (short)0, c, false, false);
}

// A operand (16x32, M x K): lane r = lane%16 is the row; this lane holds
// k = 16*khalf + i (i=0..15), contiguous.
// rowk points at &A[r][kc] (row-major, contiguous K). Two 16B loads.
__device__ __forceinline__ v16h load_a(const h16* rowk, int khalf) {
  v8h lo = *reinterpret_cast<const v8h*>(rowk + 16 * khalf);
  v8h hi = *reinterpret_cast<const v8h*>(rowk + 16 * khalf + 8);
  v16h a;
#pragma unroll
  for (int i = 0; i < 8; ++i) { a[i] = lo[i]; a[i + 8] = hi[i]; }
  return a;
}

// B operand (32x16, K x N): lane c = lane%16 is the column; this lane holds
// k = 16*khalf + i (i=0..15), contiguous. We keep B pre-transposed as
// BT[n][k] row-major, so btk = &BT[c][kc + 16*khalf]. Two 16B loads.
__device__ __forceinline__ v16h load_b(const h16* btk) {
  v8h lo = *reinterpret_cast<const v8h*>(btk);
  v8h hi = *reinterpret_cast<const v8h*>(btk + 8);
  v16h b;
#pragma unroll
  for (int i = 0; i < 8; ++i) { b[i] = lo[i]; b[i + 8] = hi[i]; }
  return b;
}

// Async-stage one 16x32 f16 tile of row-major A (row stride D_ halves) into
// LDS, issued by the calling wave (must be wave 0, full EXEC). 1KB total:
// 2 x (32 lanes x B128). ASYNCcnt-tracked; caller barriers afterwards.
__device__ __forceinline__ void async_stage_a_tile(h16* lds_tile,
                                                   const h16* gtile,
                                                   int lane) {
  unsigned ldsbase = (unsigned)(uintptr_t)lds_tile;     // low 32b = LDS offset
#pragma unroll
  for (int j = 0; j < 2; ++j) {
    int f = (j * 32 + lane) * 16;                       // byte index in tile
    int row = f >> 6, cb = f & 63;                      // 64B per tile row
    unsigned long long ga =
        (unsigned long long)(uintptr_t)((const char*)gtile + (size_t)row * (D_ * 2) + cb);
    unsigned la = ldsbase + (unsigned)f;
    asm volatile("global_load_async_to_lds_b128 %0, %1, off"
                 :: "v"(la), "v"(ga) : "memory");
  }
  asm volatile("s_wait_asynccnt 0" ::: "memory");
}

// ----- Kernel 1: weight transpose + f16 convert ----------------------------

__global__ void __launch_bounds__(256)
convert_weights(const float* __restrict__ Win, const float* __restrict__ Wout,
                h16* __restrict__ wtin, h16* __restrict__ wtout) {
  int idx = blockIdx.x * blockDim.x + threadIdx.x;
  if (idx < D3_ * D_) {                 // wtin[n][k] = Win[k][n]
    int n = idx / D_, k = idx % D_;
    wtin[idx] = (h16)Win[(size_t)k * D3_ + n];
  }
  if (idx < D_ * D_) {                  // wtout[n][k] = Wout[k][n]
    int n = idx / D_, k = idx % D_;
    wtout[idx] = (h16)Wout[(size_t)k * D_ + n];
  }
}

// ----- Kernel 2: pre-LayerNorm, emit f16 -----------------------------------

__global__ void __launch_bounds__(256)
ln_kernel(const float* __restrict__ x, const float* __restrict__ gamma,
          const float* __restrict__ beta, h16* __restrict__ out) {
  __shared__ float red[8];
  int row = blockIdx.x;                       // 0..BL-1
  const float* xr = x + (size_t)row * D_;
  float s = 0.f;
  for (int i = threadIdx.x; i < D_; i += 256) s += xr[i];
#pragma unroll
  for (int off = 16; off; off >>= 1) s += __shfl_xor(s, off, 32);
  if ((threadIdx.x & 31) == 0) red[threadIdx.x >> 5] = s;
  __syncthreads();
  float mean = 0.f;
#pragma unroll
  for (int w = 0; w < 8; ++w) mean += red[w];
  mean *= (1.0f / D_);
  __syncthreads();
  float v = 0.f;
  for (int i = threadIdx.x; i < D_; i += 256) {
    float d = xr[i] - mean; v += d * d;
  }
#pragma unroll
  for (int off = 16; off; off >>= 1) v += __shfl_xor(v, off, 32);
  if ((threadIdx.x & 31) == 0) red[threadIdx.x >> 5] = v;
  __syncthreads();
  float var = 0.f;
#pragma unroll
  for (int w = 0; w < 8; ++w) var += red[w];
  var *= (1.0f / D_);
  float inv = rsqrtf(var + 1e-6f);
  for (int i = threadIdx.x; i < D_; i += 256)
    out[(size_t)row * D_ + i] = (h16)((xr[i] - mean) * inv * gamma[i] + beta[i]);
}

// ----- Kernel 3: fused QKV GEMM (4096x3072, K=1024) + head scatter ---------
// Block: one 16-row M tile x 512 columns; wave w owns columns w*64..w*64+63.
// A tile shared through LDS (async-staged); V written transposed [B,H,dh,L].

__global__ void __launch_bounds__(256)
qkv_gemm(const h16* __restrict__ A, const h16* __restrict__ BT,
         const float* __restrict__ bias, h16* __restrict__ qh,
         h16* __restrict__ kh, h16* __restrict__ vt) {
  __shared__ __align__(16) h16 a_lds[16][32];
  const int wave = threadIdx.x >> 5, lane = threadIdx.x & 31;
  const int khalf = lane >> 4, lc = lane & 15;
  const int NG = D3_ / 512;                      // 6 column groups
  const int mt = blockIdx.x / NG;
  const int n0 = (blockIdx.x % NG) * 512 + wave * 64;

  const h16* abase = A + (size_t)(mt * 16) * D_;
  const h16* brow0 = BT + (size_t)(n0 + lc) * D_;         // j-th subtile: +16j rows
  v8f acc0 = {}, acc1 = {}, acc2 = {}, acc3 = {};

  for (int kc = 0; kc < D_; kc += 32) {
    __syncthreads();                              // previous tile fully read
    if (wave == 0) async_stage_a_tile(&a_lds[0][0], abase + kc, lane);
    __syncthreads();                              // tile published
    v16h a = load_a(&a_lds[lc][0], khalf);
    acc0 = wmma32(a, load_b(brow0 + (size_t)0  * D_ + kc + 16 * khalf), acc0);
    acc1 = wmma32(a, load_b(brow0 + (size_t)16 * D_ + kc + 16 * khalf), acc1);
    acc2 = wmma32(a, load_b(brow0 + (size_t)32 * D_ + kc + 16 * khalf), acc2);
    acc3 = wmma32(a, load_b(brow0 + (size_t)48 * D_ + kc + 16 * khalf), acc3);
  }

  v8f accs[4] = {acc0, acc1, acc2, acc3};
#pragma unroll
  for (int j = 0; j < 4; ++j) {
    const int n = n0 + j * 16 + lc;              // output column (C layout)
    const float bv = bias[n];
    const int which = n >> 10;                   // 0:q 1:k 2:v (uniform per j)
    const int nd = n & 1023, h = nd >> 6, d = nd & 63;
#pragma unroll
    for (int i = 0; i < 8; ++i) {
      int m = mt * 16 + i + 8 * khalf;           // global row = b*L + l
      int bb = m >> 11, l = m & (L_ - 1);
      h16 hv = (h16)(accs[j][i] + bv);
      size_t hb = (size_t)bb * H_ + h;
      if (which == 0)      qh[(hb * L_ + l) * DH_ + d] = hv;
      else if (which == 1) kh[(hb * L_ + l) * DH_ + d] = hv;
      else                 vt[(hb * DH_ + d) * L_ + l] = hv;
    }
  }
}

// ----- Kernel 4: flash attention, one wave per 16-query tile ---------------

__global__ void __launch_bounds__(256)
flash_attn(const h16* __restrict__ qh, const h16* __restrict__ kh,
           const h16* __restrict__ vt, const unsigned char* __restrict__ mask,
           h16* __restrict__ ctx) {
  __shared__ __align__(16) h16 lds_p[8][16][32];   // per-wave P staging (8KB)
  const int wave = threadIdx.x >> 5, lane = threadIdx.x & 31;
  const int khalf = lane >> 4, lc = lane & 15;

  int tile = blockIdx.x * 8 + wave;                // 4096 tiles
  int qt = tile & 127, h = (tile >> 7) & 15, bb = tile >> 11;

  size_t hb = (size_t)bb * H_ + h;
  const h16* qbase = qh + (hb * L_ + qt * 16) * DH_;
  const h16* kbase = kh + hb * L_ * DH_;
  const h16* vbase = vt + hb * DH_ * L_;
  const unsigned char* mrow = mask + ((size_t)bb * L_ + qt * 16) * L_;

  // Q operand resident in regs: row = lc, two dh-chunks of 32
  v16h aq0 = load_a(qbase + (size_t)lc * DH_ + 0,  khalf);
  v16h aq1 = load_a(qbase + (size_t)lc * DH_ + 32, khalf);

  v8f o0 = {}, o1 = {}, o2 = {}, o3 = {};
  float m_[8], l_[8];
#pragma unroll
  for (int i = 0; i < 8; ++i) { m_[i] = -3.0e38f; l_[i] = 0.f; }
  const float scale = 0.125f;                      // 1/sqrt(64)

  for (int kt = 0; kt < L_; kt += 32) {
    // prefetch next key/value tiles (global_prefetch_b8)
    if (kt + 32 < L_) {
      __builtin_prefetch(kbase + (size_t)(kt + 32 + lc) * DH_, 0, 0);
      __builtin_prefetch(vbase + (size_t)lc * L_ + kt + 32, 0, 0);
    }
    // --- S = Q K^T for 32 keys (two 16-key column tiles) ---
    v8f s0 = {}, s1 = {};
    {
      const h16* k0 = kbase + (size_t)(kt + lc) * DH_ + 16 * khalf;
      s0 = wmma32(aq0, load_b(k0), s0);
      s0 = wmma32(aq1, load_b(k0 + 32), s0);
      const h16* k1 = kbase + (size_t)(kt + 16 + lc) * DH_ + 16 * khalf;
      s1 = wmma32(aq0, load_b(k1), s1);
      s1 = wmma32(aq1, load_b(k1 + 32), s1);
    }
    // --- mask + online softmax (row across 16 lanes of each half-wave) ---
    float p0[8], p1[8], alpha[8];
#pragma unroll
    for (int i = 0; i < 8; ++i) {
      int q = i + 8 * khalf;
      const unsigned char* mq = mrow + (size_t)q * L_;
      float v0 = mq[kt + lc]      ? -1.0e30f : s0[i] * scale;
      float v1 = mq[kt + 16 + lc] ? -1.0e30f : s1[i] * scale;
      float mx = fmaxf(v0, v1);
#pragma unroll
      for (int off = 8; off; off >>= 1) mx = fmaxf(mx, __shfl_xor(mx, off, 32));
      float mn = fmaxf(m_[i], mx);
      float al = __expf(m_[i] - mn);
      float e0 = __expf(v0 - mn), e1 = __expf(v1 - mn);
      float rs = e0 + e1;
#pragma unroll
      for (int off = 8; off; off >>= 1) rs += __shfl_xor(rs, off, 32);
      l_[i] = l_[i] * al + rs;
      m_[i] = mn; alpha[i] = al; p0[i] = e0; p1[i] = e1;
    }
#pragma unroll
    for (int i = 0; i < 8; ++i) {
      o0[i] *= alpha[i]; o1[i] *= alpha[i];
      o2[i] *= alpha[i]; o3[i] *= alpha[i];
    }
    // --- C-layout -> A-layout transpose of P through LDS (same-wave DS) ---
#pragma unroll
    for (int i = 0; i < 8; ++i) {
      int q = i + 8 * khalf;
      lds_p[wave][q][lc]      = (h16)p0[i];
      lds_p[wave][q][16 + lc] = (h16)p1[i];
    }
    v16h pa = load_a(&lds_p[wave][lc][0], khalf);
    // --- O += P V  (V pre-transposed: vbase[d][key], contiguous key) ---
    o0 = wmma32(pa, load_b(vbase + (size_t)(lc)      * L_ + kt + 16 * khalf), o0);
    o1 = wmma32(pa, load_b(vbase + (size_t)(16 + lc) * L_ + kt + 16 * khalf), o1);
    o2 = wmma32(pa, load_b(vbase + (size_t)(32 + lc) * L_ + kt + 16 * khalf), o2);
    o3 = wmma32(pa, load_b(vbase + (size_t)(48 + lc) * L_ + kt + 16 * khalf), o3);
  }
  // --- epilogue: normalize and scatter ctx as [B, L, H*dh] ---
#pragma unroll
  for (int i = 0; i < 8; ++i) {
    int q = qt * 16 + i + 8 * khalf;
    float inv = 1.0f / l_[i];
    size_t base = ((size_t)bb * L_ + q) * D_ + h * DH_;
    ctx[base + lc]      = (h16)(o0[i] * inv);
    ctx[base + 16 + lc] = (h16)(o1[i] * inv);
    ctx[base + 32 + lc] = (h16)(o2[i] * inv);
    ctx[base + 48 + lc] = (h16)(o3[i] * inv);
  }
}

// ----- Kernel 5: output projection (4096x1024, K=1024) + bias + residual ---

__global__ void __launch_bounds__(256)
out_gemm(const h16* __restrict__ A, const h16* __restrict__ BT,
         const float* __restrict__ bias, const float* __restrict__ resid,
         float* __restrict__ out) {
  __shared__ __align__(16) h16 a_lds[16][32];
  const int wave = threadIdx.x >> 5, lane = threadIdx.x & 31;
  const int khalf = lane >> 4, lc = lane & 15;
  const int NG = D_ / 512;                       // 2 column groups
  const int mt = blockIdx.x / NG;
  const int n0 = (blockIdx.x % NG) * 512 + wave * 64;

  const h16* abase = A + (size_t)(mt * 16) * D_;
  const h16* brow0 = BT + (size_t)(n0 + lc) * D_;
  v8f acc0 = {}, acc1 = {}, acc2 = {}, acc3 = {};

  for (int kc = 0; kc < D_; kc += 32) {
    __syncthreads();
    if (wave == 0) async_stage_a_tile(&a_lds[0][0], abase + kc, lane);
    __syncthreads();
    v16h a = load_a(&a_lds[lc][0], khalf);
    acc0 = wmma32(a, load_b(brow0 + (size_t)0  * D_ + kc + 16 * khalf), acc0);
    acc1 = wmma32(a, load_b(brow0 + (size_t)16 * D_ + kc + 16 * khalf), acc1);
    acc2 = wmma32(a, load_b(brow0 + (size_t)32 * D_ + kc + 16 * khalf), acc2);
    acc3 = wmma32(a, load_b(brow0 + (size_t)48 * D_ + kc + 16 * khalf), acc3);
  }

  v8f accs[4] = {acc0, acc1, acc2, acc3};
#pragma unroll
  for (int j = 0; j < 4; ++j) {
    const int n = n0 + j * 16 + lc;
    const float bv = bias[n];
#pragma unroll
    for (int i = 0; i < 8; ++i) {
      size_t m = (size_t)(mt * 16 + i + 8 * khalf);
      out[m * D_ + n] = accs[j][i] + bv + resid[m * D_ + n];
    }
  }
}

// ----- Host launcher -------------------------------------------------------

extern "C" void kernel_launch(void* const* d_in, const int* in_sizes, int n_in,
                              void* d_out, int out_size, void* d_ws, size_t ws_size,
                              hipStream_t stream) {
  const float* x     = (const float*)d_in[0];            // [B,L,D]
  const unsigned char* mask = (const unsigned char*)d_in[1]; // [B,L,L] bool
  const float* Win   = (const float*)d_in[2];            // [D,3D]
  const float* bin   = (const float*)d_in[3];            // [3D]
  const float* Wout  = (const float*)d_in[4];            // [D,D]
  const float* bout  = (const float*)d_in[5];            // [D]
  const float* gamma = (const float*)d_in[6];
  const float* beta  = (const float*)d_in[7];
  float* out = (float*)d_out;

  // Workspace carve-up (all f16), ~48 MB total.
  char* ws = (char*)d_ws;
  size_t off = 0;
  auto take = [&](size_t bytes) { char* p = ws + off; off += (bytes + 255) & ~(size_t)255; return p; };
  h16* normed = (h16*)take((size_t)BL_ * D_  * 2);   // 8 MB
  h16* wtin   = (h16*)take((size_t)D3_ * D_  * 2);   // 6 MB
  h16* wtout  = (h16*)take((size_t)D_  * D_  * 2);   // 2 MB
  h16* qh     = (h16*)take((size_t)B_ * H_ * L_ * DH_ * 2); // 8 MB
  h16* kh     = (h16*)take((size_t)B_ * H_ * L_ * DH_ * 2); // 8 MB
  h16* vt     = (h16*)take((size_t)B_ * H_ * DH_ * L_ * 2); // 8 MB
  h16* ctx    = (h16*)take((size_t)BL_ * D_  * 2);   // 8 MB
  (void)ws_size; (void)n_in; (void)in_sizes; (void)out_size;

  // 1) weights -> transposed f16
  convert_weights<<<(D3_ * D_ + 255) / 256, 256, 0, stream>>>(Win, Wout, wtin, wtout);
  // 2) pre-LN
  ln_kernel<<<BL_, 256, 0, stream>>>(x, gamma, beta, normed);
  // 3) QKV projection: 256 M-tiles x 6 column groups
  qkv_gemm<<<(BL_ / 16) * (D3_ / 512), 256, 0, stream>>>(normed, wtin, bin, qh, kh, vt);
  // 4) flash attention (4096 wave-tiles)
  flash_attn<<<B_ * H_ * (L_ / 16) / 8, 256, 0, stream>>>(qh, kh, vt, mask, ctx);
  // 5) output projection + bias + residual
  out_gemm<<<(BL_ / 16) * (D_ / 512), 256, 0, stream>>>(ctx, wtout, bout, x, out);
}
